// EncoderLayer_66460323938534
// MI455X (gfx1250) — matrix operand order repaired and checked by reading, not compiled, not run on hardware
//
#include <hip/hip_runtime.h>
#include <cstddef>
#include <cstdint>

// ---------------- problem constants ----------------
#define B_   8
#define L_   2048
#define DM_  512
#define H_   8
#define DH_  64
#define S_   33
#define U_   33
#define MROWS_ (B_ * L_)          // 16384
#define EPS_ 1e-5f

typedef __attribute__((ext_vector_type(16))) __bf16 v16bf;
typedef __attribute__((ext_vector_type(8)))  float  v8f;

__device__ __forceinline__ float bf2f(__bf16 v) { return (float)v; }
__device__ __forceinline__ __bf16 f2bf(float v) { return (__bf16)v; }

// CDNA5 async global->LDS copy (ASYNCcnt path, CDNA5 ISA).
// vdst = wave-relative LDS byte address, vaddr = 64-bit global address (GV mode).
__device__ __forceinline__ void async_ld_b128(unsigned lds_off, const void* gaddr) {
    asm volatile("global_load_async_to_lds_b128 %0, %1, off"
                 :: "v"(lds_off), "v"(gaddr) : "memory");
}
__device__ __forceinline__ void wait_async0() {
    asm volatile("s_wait_asynccnt 0x0" ::: "memory");
}

// ---------------- f32 -> bf16 convert ----------------
__global__ __launch_bounds__(256) void cvt_f32_bf16(const float* __restrict__ src,
                                                    __bf16* __restrict__ dst, int n) {
    int i = blockIdx.x * 256 + threadIdx.x;
    if (i < n) dst[i] = f2bf(src[i]);
}

// weight convert + transpose: dst[n*DM_+k] = bf16(src[k*DM_+n])  (N x K)
__global__ __launch_bounds__(256) void cvt_w_bf16_t(const float* __restrict__ src,
                                                    __bf16* __restrict__ dst) {
    int i = blockIdx.x * 256 + threadIdx.x;
    if (i >= DM_ * DM_) return;
    const int n = i >> 9, k = i & (DM_ - 1);
    dst[i] = f2bf(src[(size_t)k * DM_ + n]);
}

// ---------------- WMMA GEMM ----------------
// C[M,N] = A[M,K](bf16 row-major) @ B[K,N] + bias, with B given TRANSPOSED (Bt = N x K)
// mode 0: store bf16 to (B,H,L,DH) heads layout
// mode 1: store f32 row-major
// mode 2: relu then store bf16 row-major
#define BM 128
#define BN 128
#define BK 32

__global__ __launch_bounds__(256)
void gemm_wmma_bf16(const __bf16* __restrict__ A, const __bf16* __restrict__ Bt,
                    const float* __restrict__ bias, void* __restrict__ Cout,
                    int M, int N, int K, int mode) {
    // double-buffered tiles, both stored k-contiguous: lA[m][k], lB[n][k]
    __shared__ alignas(16) __bf16 lA[2][BM * BK];
    __shared__ alignas(16) __bf16 lB[2][BN * BK];

    const int tid  = threadIdx.x;
    const int wave = tid >> 5;
    const int lane = tid & 31;
    const int l16  = lane & 15;
    const int lhi  = lane >> 4;

    const int bm = blockIdx.x * BM;
    const int bn = blockIdx.y * BN;

    // 8 waves: mw = wave&3 -> 32-row group, nw = wave>>2 -> 64-col group
    // each wave computes 2 (M) x 4 (N) subtiles of 16x16 -> 8 WMMA / 6 frag loads
    const int mw = wave & 3;
    const int nw = wave >> 2;

    v8f acc[2][4];
#pragma unroll
    for (int i = 0; i < 2; ++i)
#pragma unroll
        for (int j = 0; j < 4; ++j) acc[i][j] = (v8f){};

    // staging: tile is BM(=BN) x BK bf16 = 8KB = 512 x 16B chunks; 2 chunks/thread
    const int c0 = tid, c1 = tid + 256;
    const int rA0 = c0 >> 2, cA0 = (c0 & 3) * 8;
    const int rA1 = c1 >> 2, cA1 = (c1 & 3) * 8;

    auto issue_tile = [&](int k0, int buf) {
        async_ld_b128((unsigned)(size_t)(const void*)&lA[buf][c0 * 8],
                      A  + (size_t)(bm + rA0) * K + (k0 + cA0));
        async_ld_b128((unsigned)(size_t)(const void*)&lA[buf][c1 * 8],
                      A  + (size_t)(bm + rA1) * K + (k0 + cA1));
        async_ld_b128((unsigned)(size_t)(const void*)&lB[buf][c0 * 8],
                      Bt + (size_t)(bn + rA0) * K + (k0 + cA0));
        async_ld_b128((unsigned)(size_t)(const void*)&lB[buf][c1 * 8],
                      Bt + (size_t)(bn + rA1) * K + (k0 + cA1));
    };

    issue_tile(0, 0);
    wait_async0();
    __syncthreads();

    int buf = 0;
    for (int k0 = 0; k0 < K; k0 += BK, buf ^= 1) {
        const bool more = (k0 + BK) < K;
        if (more) issue_tile(k0 + BK, buf ^ 1);   // overlaps with WMMA below

        union Frag { v16bf v; unsigned u[8]; };
        Frag fa[2], fb[4];
#pragma unroll
        for (int i = 0; i < 2; ++i) {
            const int mrow = (mw * 2 + i) * 16 + l16;
#pragma unroll
            for (int j = 0; j < 8; ++j) {
                // A 16x32 bf16 frag: K = (j/4)*16 + (lane/16)*8 + (j%2-pair)
                const int kk = ((j >> 2) << 4) + (lhi << 3) + ((j & 3) << 1);
                fa[i].u[j] = *reinterpret_cast<const unsigned*>(&lA[buf][mrow * BK + kk]);
            }
        }
#pragma unroll
        for (int i = 0; i < 4; ++i) {
            const int nrow = (nw * 4 + i) * 16 + l16;
#pragma unroll
            for (int j = 0; j < 8; ++j) {
                // B 32x16 bf16 frag: K = (j/4)*16 + (lane/16)*8 + (j%2-pair), lane%16 = N
                const int kk = ((j >> 2) << 4) + (lhi << 3) + ((j & 3) << 1);
                fb[i].u[j] = *reinterpret_cast<const unsigned*>(&lB[buf][nrow * BK + kk]);
            }
        }
#pragma unroll
        for (int i = 0; i < 2; ++i)
#pragma unroll
            for (int j = 0; j < 4; ++j)
                acc[i][j] = __builtin_amdgcn_wmma_f32_16x16x32_bf16(
                    false, fa[i].v, false, fb[j].v, (short)0, acc[i][j], false, false);

        wait_async0();        // own async writes for next tile done
        __syncthreads();      // all waves: done reading buf, next buf fully written
    }

    // ---- epilogue ----
#pragma unroll
    for (int i = 0; i < 2; ++i) {
#pragma unroll
        for (int j = 0; j < 4; ++j) {
            const int gn = bn + (nw * 4 + j) * 16 + l16;
            const float bv = bias[gn];
#pragma unroll
            for (int r = 0; r < 8; ++r) {
                const int gm = bm + (mw * 2 + i) * 16 + (lhi << 3) + r;
                float val = acc[i][j][r] + bv;
                if (mode == 0) {
                    const int b = gm >> 11, l = gm & (L_ - 1);
                    const int h = gn >> 6,  d = gn & (DH_ - 1);
                    ((__bf16*)Cout)[(((size_t)b * H_ + h) * L_ + l) * DH_ + d] = f2bf(val);
                } else if (mode == 1) {
                    ((float*)Cout)[(size_t)gm * N + gn] = val;
                } else {
                    if (val < 0.f) val = 0.f;
                    ((__bf16*)Cout)[(size_t)gm * N + gn] = f2bf(val);
                }
            }
        }
    }
}

// ---------------- sampled qk -> M = max - mean ----------------
__global__ __launch_bounds__(256)
void sampled_qk(const __bf16* __restrict__ qb, const __bf16* __restrict__ kb,
                const int* __restrict__ sidx, float* __restrict__ Mout) {
    const int gid = blockIdx.x * 256 + threadIdx.x;     // (bh, l)
    if (gid >= B_ * H_ * L_) return;
    const int l  = gid & (L_ - 1);
    const int bh = gid >> 11;
    const __bf16* qrow = qb + ((size_t)bh * L_ + l) * DH_;
    float qf[DH_];
#pragma unroll
    for (int d = 0; d < DH_; ++d) qf[d] = bf2f(qrow[d]);
    float mx = -3.4e38f, sum = 0.f;
    for (int s = 0; s < S_; ++s) {
        const int idx = sidx[l * S_ + s];
        const __bf16* krow = kb + ((size_t)bh * L_ + idx) * DH_;
        float dot = 0.f;
#pragma unroll
        for (int d = 0; d < DH_; ++d) dot += qf[d] * bf2f(krow[d]);
        mx = fmaxf(mx, dot);
        sum += dot;
    }
    Mout[gid] = mx - sum / (float)S_;
}

// ---------------- top-U per (b,h): iterative argmax ----------------
__global__ __launch_bounds__(256)
void topk_kernel(const float* __restrict__ Mbuf, int* __restrict__ topidx) {
    __shared__ float sv[L_];
    __shared__ float rv[256];
    __shared__ int   ri[256];
    const int bh = blockIdx.x;
    const int t  = threadIdx.x;
    for (int i = t; i < L_; i += 256) sv[i] = Mbuf[(size_t)bh * L_ + i];
    __syncthreads();
    for (int u = 0; u < U_; ++u) {
        float best = -3.4e38f;
        int   bi   = 0x7fffffff;
        for (int i = t; i < L_; i += 256) {
            const float x = sv[i];
            if (x > best || (x == best && i < bi)) { best = x; bi = i; }
        }
        rv[t] = best; ri[t] = bi;
        __syncthreads();
        for (int s = 128; s > 0; s >>= 1) {
            if (t < s) {
                if (rv[t + s] > rv[t] || (rv[t + s] == rv[t] && ri[t + s] < ri[t])) {
                    rv[t] = rv[t + s]; ri[t] = ri[t + s];
                }
            }
            __syncthreads();
        }
        if (t == 0) { topidx[bh * U_ + u] = ri[0]; sv[ri[0]] = -3.4e38f; }
        __syncthreads();
    }
}

// ---------------- full attention row for each (b,h,u) ----------------
__global__ __launch_bounds__(256)
void attn_topk(const __bf16* __restrict__ qb, const __bf16* __restrict__ kb,
               const __bf16* __restrict__ vb, const int* __restrict__ topidx,
               float* __restrict__ outtop) {
    __shared__ float qs[DH_];
    __shared__ float sc[L_];
    __shared__ float red[256];
    const int blk = blockIdx.x;
    const int u   = blk % U_;
    const int bh  = blk / U_;
    const int t   = threadIdx.x;
    const int ltop = topidx[bh * U_ + u];
    if (t < DH_) qs[t] = bf2f(qb[((size_t)bh * L_ + ltop) * DH_ + t]);
    __syncthreads();
    const float scale = 0.125f;   // 1/sqrt(64)
    for (int l = t; l < L_; l += 256) {
        const __bf16* krow = kb + ((size_t)bh * L_ + l) * DH_;
        float dot = 0.f;
#pragma unroll
        for (int d = 0; d < DH_; ++d) dot += qs[d] * bf2f(krow[d]);
        sc[l] = dot * scale;
    }
    __syncthreads();
    float mx = -3.4e38f;
    for (int l = t; l < L_; l += 256) mx = fmaxf(mx, sc[l]);
    red[t] = mx; __syncthreads();
    for (int s = 128; s > 0; s >>= 1) { if (t < s) red[t] = fmaxf(red[t], red[t + s]); __syncthreads(); }
    mx = red[0]; __syncthreads();
    float sm = 0.f;
    for (int l = t; l < L_; l += 256) { const float e = __expf(sc[l] - mx); sc[l] = e; sm += e; }
    red[t] = sm; __syncthreads();
    for (int s = 128; s > 0; s >>= 1) { if (t < s) red[t] += red[t + s]; __syncthreads(); }
    const float denom = red[0]; __syncthreads();
    const int d    = t & (DH_ - 1);
    const int part = t >> 6;               // 0..3, each covers L/4
    float acc = 0.f;
    const int l0 = part * (L_ / 4);
    for (int l = l0; l < l0 + (L_ / 4); ++l)
        acc += sc[l] * bf2f(vb[((size_t)bh * L_ + l) * DH_ + d]);
    red[t] = acc; __syncthreads();
    if (part == 0) {
        const float s = red[d] + red[d + 64] + red[d + 128] + red[d + 192];
        outtop[((size_t)bh * U_ + u) * DH_ + d] = s / denom;
    }
}

// ---------------- v mean over L ----------------
__global__ __launch_bounds__(256)
void vmean_kernel(const __bf16* __restrict__ vb, float* __restrict__ vmean) {
    const int i = blockIdx.x * 256 + threadIdx.x;   // (bh, d)
    if (i >= B_ * H_ * DH_) return;
    const int d  = i & (DH_ - 1);
    const int bh = i >> 6;
    float s = 0.f;
    for (int l = 0; l < L_; ++l) s += bf2f(vb[((size_t)bh * L_ + l) * DH_ + d]);
    vmean[i] = s * (1.f / (float)L_);
}

// ---------------- broadcast ctx = mean(v), bf16 (B,L,DM) ----------------
__global__ __launch_bounds__(256)
void ctx_fill(const float* __restrict__ vmean, __bf16* __restrict__ ctx) {
    const int i = blockIdx.x * 256 + threadIdx.x;
    if (i >= B_ * L_ * DM_) return;
    const int col = i & (DM_ - 1);
    const int b   = i >> 20;                 // / (L_*DM_)
    const int h   = col >> 6, d = col & (DH_ - 1);
    ctx[i] = f2bf(vmean[((size_t)b * H_ + h) * DH_ + d]);
}

// ---------------- scatter out_top rows into ctx ----------------
__global__ __launch_bounds__(256)
void ctx_scatter(const float* __restrict__ outtop, const int* __restrict__ topidx,
                 __bf16* __restrict__ ctx) {
    const int i = blockIdx.x * 256 + threadIdx.x;   // (bh,u,d)
    if (i >= B_ * H_ * U_ * DH_) return;
    const int d  = i & (DH_ - 1);
    const int u  = (i >> 6) % U_;
    const int bh = (i >> 6) / U_;
    const int b  = bh >> 3, h = bh & (H_ - 1);
    const int l  = topidx[bh * U_ + u];
    ctx[((size_t)b * L_ + l) * DM_ + h * DH_ + d] =
        f2bf(outtop[((size_t)bh * U_ + u) * DH_ + d]);
}

// ---------------- fused residual + layernorm ----------------
__global__ __launch_bounds__(256)
void layernorm_res(const float* __restrict__ a, const float* __restrict__ b,
                   const float* __restrict__ g, const float* __restrict__ be,
                   float* __restrict__ outf, __bf16* __restrict__ outb) {
    __shared__ float red[256];
    const int row = blockIdx.x;
    const int t   = threadIdx.x;
    const float* pa = a + (size_t)row * DM_;
    const float* pb = b + (size_t)row * DM_;
    const float x0 = pa[t] + pb[t];
    const float x1 = pa[t + 256] + pb[t + 256];
    red[t] = x0 + x1; __syncthreads();
    for (int s = 128; s > 0; s >>= 1) { if (t < s) red[t] += red[t + s]; __syncthreads(); }
    const float mean = red[0] * (1.f / (float)DM_); __syncthreads();
    const float d0 = x0 - mean, d1 = x1 - mean;
    red[t] = d0 * d0 + d1 * d1; __syncthreads();
    for (int s = 128; s > 0; s >>= 1) { if (t < s) red[t] += red[t + s]; __syncthreads(); }
    const float var = red[0] * (1.f / (float)DM_);
    const float inv = rsqrtf(var + EPS_);
    const float y0 = d0 * inv * g[t]       + be[t];
    const float y1 = d1 * inv * g[t + 256] + be[t + 256];
    outf[(size_t)row * DM_ + t]       = y0;
    outf[(size_t)row * DM_ + t + 256] = y1;
    if (outb) {
        outb[(size_t)row * DM_ + t]       = f2bf(y0);
        outb[(size_t)row * DM_ + t + 256] = f2bf(y1);
    }
}

// ---------------- host orchestration ----------------
extern "C" void kernel_launch(void* const* d_in, const int* in_sizes, int n_in,
                              void* d_out, int out_size, void* d_ws, size_t ws_size,
                              hipStream_t stream) {
    (void)in_sizes; (void)n_in; (void)out_size; (void)ws_size;
    const float* x    = (const float*)d_in[0];
    const int*   sidx = (const int*)  d_in[1];
    const float* Wf[6] = { (const float*)d_in[2],  (const float*)d_in[4],
                           (const float*)d_in[6],  (const float*)d_in[8],
                           (const float*)d_in[10], (const float*)d_in[12] };
    const float* bq = (const float*)d_in[3];
    const float* bk = (const float*)d_in[5];
    const float* bv = (const float*)d_in[7];
    const float* bo = (const float*)d_in[9];
    const float* b1 = (const float*)d_in[11];
    const float* b2 = (const float*)d_in[13];
    const float* g1 = (const float*)d_in[14];
    const float* be1 = (const float*)d_in[15];
    const float* g2 = (const float*)d_in[16];
    const float* be2 = (const float*)d_in[17];

    char* ws = (char*)d_ws;
    const size_t szXB   = (size_t)MROWS_ * DM_ * 2;          // 16 MB
    const size_t szW    = (size_t)DM_ * DM_ * 2;             // 0.5 MB each
    const size_t szHead = (size_t)B_ * H_ * L_ * DH_ * 2;    // 16 MB
    size_t off = 0;
    __bf16* xb   = (__bf16*)(ws + off); off += szXB;
    __bf16* wb[6];
    for (int i = 0; i < 6; ++i) { wb[i] = (__bf16*)(ws + off); off += szW; }
    __bf16* qb   = (__bf16*)(ws + off); off += szHead;
    __bf16* kb   = (__bf16*)(ws + off); off += szHead;
    __bf16* vb   = (__bf16*)(ws + off); off += szHead;
    float*  Mbuf = (float*)(ws + off);  off += (size_t)B_ * H_ * L_ * 4;
    int*    tidx = (int*)(ws + off);    off += 16384;
    float*  otop = (float*)(ws + off);  off += (size_t)B_ * H_ * U_ * DH_ * 4;
    float*  vmn  = (float*)(ws + off);  off += 16384;
    float*  attn = (float*)(ws + off);  off += (size_t)MROWS_ * DM_ * 4;  // 32 MB
    float*  h1   = (float*)(ws + off);  off += (size_t)MROWS_ * DM_ * 4;  // 32 MB
    // aliases (lifetimes disjoint, stream-ordered)
    __bf16* ctx  = xb;     // x bf16 dead after QKV GEMMs
    __bf16* h1b  = qb;     // q dead after attention
    __bf16* relu = kb;     // k dead after attention
    float*  ff2  = attn;   // attn_out consumed by LN1 before W2 writes

    const dim3 blk(256);
    const dim3 gGemm(MROWS_ / BM, DM_ / BN);   // (128, 4)

    // 1. precision conversion (+ weight transpose to N x K for async LDS staging)
    cvt_f32_bf16<<<(B_ * L_ * DM_) / 256, blk, 0, stream>>>(x, xb, B_ * L_ * DM_);
    for (int i = 0; i < 6; ++i)
        cvt_w_bf16_t<<<(DM_ * DM_) / 256, blk, 0, stream>>>(Wf[i], wb[i]);

    // 2. Q/K/V projections (WMMA, heads layout)
    gemm_wmma_bf16<<<gGemm, blk, 0, stream>>>(xb, wb[0], bq, qb, MROWS_, DM_, DM_, 0);
    gemm_wmma_bf16<<<gGemm, blk, 0, stream>>>(xb, wb[1], bk, kb, MROWS_, DM_, DM_, 0);
    gemm_wmma_bf16<<<gGemm, blk, 0, stream>>>(xb, wb[2], bv, vb, MROWS_, DM_, DM_, 0);

    // 3. sampled scores -> M, top-U, attention for selected queries
    sampled_qk<<<(B_ * H_ * L_) / 256, blk, 0, stream>>>(qb, kb, sidx, Mbuf);
    topk_kernel<<<B_ * H_, blk, 0, stream>>>(Mbuf, tidx);
    attn_topk<<<B_ * H_ * U_, blk, 0, stream>>>(qb, kb, vb, tidx, otop);

    // 4. ctx = broadcast(mean v) with out_top scattered at top indices
    vmean_kernel<<<(B_ * H_ * DH_ + 255) / 256, blk, 0, stream>>>(vb, vmn);
    ctx_fill<<<(B_ * L_ * DM_) / 256, blk, 0, stream>>>(vmn, ctx);
    ctx_scatter<<<(B_ * H_ * U_ * DH_ + 255) / 256, blk, 0, stream>>>(otop, tidx, ctx);

    // 5. output projection + LN1
    gemm_wmma_bf16<<<gGemm, blk, 0, stream>>>(ctx, wb[3], bo, attn, MROWS_, DM_, DM_, 1);
    layernorm_res<<<MROWS_, blk, 0, stream>>>(x, attn, g1, be1, h1, h1b);

    // 6. FFN + LN2
    gemm_wmma_bf16<<<gGemm, blk, 0, stream>>>(h1b, wb[4], b1, relu, MROWS_, DM_, DM_, 2);
    gemm_wmma_bf16<<<gGemm, blk, 0, stream>>>(relu, wb[5], b2, ff2, MROWS_, DM_, DM_, 1);
    layernorm_res<<<MROWS_, blk, 0, stream>>>(h1, ff2, g2, be2, (float*)d_out, (__bf16*)nullptr);
}
